// TransIKGDecoder_31293131719319
// MI455X (gfx1250) — compile-verified, matrix operand order checked by reading, not run on hardware
//
#include <hip/hip_runtime.h>
#include <hip/hip_bf16.h>

typedef __attribute__((ext_vector_type(16))) __bf16 v16bf;
typedef __attribute__((ext_vector_type(8)))  float  v8f;
typedef __attribute__((ext_vector_type(4)))  unsigned int v4u;
typedef __attribute__((ext_vector_type(8)))  int    v8i;
typedef __attribute__((ext_vector_type(4)))  int    v4i;

#define NEG_INF (-1e20f)

// ---------------------------------------------------------------------------
// WMMA helper: D = A(16x32 bf16) * B(32x16 bf16) + C(16x16 f32)
// ---------------------------------------------------------------------------
__device__ __forceinline__ v8f wmma_bf16(v16bf a, v16bf b, v8f c) {
  return __builtin_amdgcn_wmma_f32_16x16x32_bf16(
      /*neg_a=*/false, a, /*neg_b=*/false, b,
      /*c_mod=*/(short)0, c, /*reuse_a=*/false, /*reuse_b=*/false);
}

// Load a 16x32 bf16 tile (row-major, leading dim ld) into the CDNA5 16-bit
// A-matrix register layout (lane l: row l&15, K = {8hi..}u{16+8hi..}).
// B-matrix layout mirrors this with lane = column, so the same loader reads
// the row-major transposed operand (rows of W / rows of K).
__device__ __forceinline__ v16bf load_tile(const __bf16* p, int ld) {
  const int lane = threadIdx.x & 31;
  const int r = lane & 15, hi = lane >> 4;
  const __bf16* row = p + (size_t)r * ld + hi * 8;
  v16bf a;
#pragma unroll
  for (int i = 0; i < 8; ++i) {
    a[i]     = row[i];
    a[8 + i] = row[16 + i];
  }
  return a;
}

// ---------------------------------------------------------------------------
// Tensor Data Mover: DMA a 2-D bf16 tile (tile_h rows x tile_w elems, row
// pitch elem_stride elems) from global memory into LDS at lds_off.
// D# per CDNA5 ISA 8.3/8.4; tile described as the whole tensor (no OOB).
// Issued wave-uniform (EXEC is ignored by TDM); tracked by TENSORcnt.
// 6-arg toolchain variant: (g0 u32x4, g1 i32x8, g2 i32x4, g3 i32x4,
//                           g4 i32x8, cpol).
// ---------------------------------------------------------------------------
__device__ __forceinline__ void tdm_load_2d(unsigned int lds_off,
                                            const __bf16* gptr,
                                            int elem_stride,
                                            int tile_w, int tile_h) {
  unsigned long long ga = (unsigned long long)(uintptr_t)gptr;
  v4u g0;
  g0[0] = 1u;                                        // count=1 (valid D#)
  g0[1] = lds_off;                                   // LDS byte address
  g0[2] = (unsigned int)ga;                          // global_addr[31:0]
  g0[3] = (unsigned int)((ga >> 32) & 0x1FFFFFFu)    // global_addr[56:32]
          | (2u << 30);                              // type=2 ("image")
  v8i g1;
  g1[0] = (1 << 16);                                 // data_size=1 -> 2 bytes
  g1[1] = (int)((unsigned int)(tile_w & 0xFFFF) << 16);   // tensor_dim0 lo16
  g1[2] = (int)(((unsigned int)(tile_w >> 16) & 0xFFFF)   // tensor_dim0 hi16
          | ((unsigned int)(tile_h & 0xFFFF) << 16));     // tensor_dim1 lo16
  g1[3] = (int)(((unsigned int)(tile_h >> 16) & 0xFFFF)   // tensor_dim1 hi16
          | ((unsigned int)tile_w << 16));                // tile_dim0
  g1[4] = tile_h;                                    // tile_dim1 (tile_dim2=0)
  g1[5] = elem_stride;                               // tensor_dim0_stride lo32
  g1[6] = 0;                                         // stride hi16 | dim1_stride
  g1[7] = 0;
  v4i gz4 = {0, 0, 0, 0};
  v8i gz8 = {0, 0, 0, 0, 0, 0, 0, 0};
  __builtin_amdgcn_tensor_load_to_lds(g0, g1, gz4, gz4, gz8, 0);
}

// ---------------------------------------------------------------------------
// f32 -> bf16 elementwise conversion
// ---------------------------------------------------------------------------
__global__ void cvt_f32_bf16(const float* __restrict__ src,
                             __bf16* __restrict__ dst, int n) {
  int i = blockIdx.x * 256 + threadIdx.x;
  if (i < n) dst[i] = (__bf16)src[i];
}

// ---------------------------------------------------------------------------
// dpf[b,k] = cos-term + clamp(key[b,k,:]·Wb + bb, -0.5, 0.5)
// One wave per key row (4096 rows), shuffle reduction over D=1024.
// ---------------------------------------------------------------------------
__global__ void dpf_kernel(const float* __restrict__ key,
                           const float* __restrict__ Wb,
                           const float* __restrict__ bb,
                           const int* __restrict__ segments,
                           float* __restrict__ dpf) {
  const int lane = threadIdx.x & 31;
  const int w = threadIdx.x >> 5;
  const int row = blockIdx.x * 8 + w;          // 0..4095 = b*1024 + k
  const float* kr = key + (size_t)row * 1024;
  float s = 0.f;
#pragma unroll 8
  for (int i = lane; i < 1024; i += 32) s += kr[i] * Wb[i];
  s += __shfl_xor(s, 16, 32);
  s += __shfl_xor(s, 8, 32);
  s += __shfl_xor(s, 4, 32);
  s += __shfl_xor(s, 2, 32);
  s += __shfl_xor(s, 1, 32);
  if (lane == 0) {
    float bias = fminf(0.5f, fmaxf(-0.5f, s + bb[0]));
    float seg = (float)segments[row];
    const float m = 5.0f;                      // n_segments
    float v = (cosf(3.14159265358979323846f * seg) - (4.0f / m) * seg + 7.0f +
               (3.0f * m + 2.0f) / (m + 1.0f)) * 0.125f + bias;
    dpf[row] = v;
  }
}

// ---------------------------------------------------------------------------
// GEMM: out[m,n] = (sum_k A[m,k]*W[n,k] + bias[n]) * scale(m)
// Block = 256 thr (8 waves); block tile 128(M) x 64(N), K step 32.
// The 64x32 W tile is shared by all 8 waves: TDM-DMA'd into LDS,
// double-buffered so the DMA of step k+32 overlaps the WMMAs of step k.
// headSplit: write bf16 to [B, H=16, L, dh=64]; else write f32 [M,N].
// ---------------------------------------------------------------------------
__global__ void gemm_bf16(const __bf16* __restrict__ A,
                          const __bf16* __restrict__ W,
                          const float* __restrict__ bias,
                          const float* __restrict__ rowscale, // may be null
                          float constscale,
                          __bf16* __restrict__ outbf,
                          float* __restrict__ outf32,
                          int M, int N, int K, int headSplit, int L) {
  const int lane = threadIdx.x & 31;
  const int w = threadIdx.x >> 5;
  const int r = lane & 15, hi = lane >> 4;
  const int m0 = blockIdx.x * 128 + w * 16;
  const int n0 = blockIdx.y * 64;

  __shared__ __bf16 wtile[2][64][32];          // [buf][n rows][k cols]

  if (w == 0)                                  // prologue DMA: k0 = 0
    tdm_load_2d((unsigned int)(uintptr_t)&wtile[0][0][0],
                W + (size_t)n0 * K, K, 32, 64);

  v8f acc[4] = {};
  for (int k0 = 0; k0 < K; k0 += 32) {
    const int cur = (k0 >> 5) & 1;
    if (w == 0) {
      if (k0 + 32 < K) {
        tdm_load_2d((unsigned int)(uintptr_t)&wtile[cur ^ 1][0][0],
                    W + (size_t)n0 * K + (k0 + 32), K, 32, 64);
        __builtin_amdgcn_s_wait_tensorcnt(1);  // cur tile landed (in-order)
      } else {
        __builtin_amdgcn_s_wait_tensorcnt(0);
      }
    }
    __syncthreads();                           // publish wtile[cur]

    v16bf a = load_tile(A + (size_t)m0 * K + k0, K);
#pragma unroll
    for (int t = 0; t < 4; ++t) {
      v16bf bt = load_tile(&wtile[cur][t * 16][0], 32);
      acc[t] = wmma_bf16(a, bt, acc[t]);
    }
    __syncthreads();                           // done reading wtile[cur]
  }

#pragma unroll
  for (int t = 0; t < 4; ++t) {
#pragma unroll
    for (int j = 0; j < 8; ++j) {
      const int m = m0 + hi * 8 + j;           // C layout: (M=8hi+j, N=r)
      const int n = n0 + t * 16 + r;
      float v = acc[t][j] + bias[n];
      v *= (rowscale ? rowscale[m] : constscale);
      if (headSplit) {
        const int b = m / L, l = m % L, h = n >> 6, d = n & 63;
        outbf[(((size_t)b * 16 + h) * L + l) * 64 + d] = (__bf16)v;
      } else {
        outf32[(size_t)m * N + n] = v;
      }
    }
  }
}

// ---------------------------------------------------------------------------
// Flash attention (online softmax). One wave owns 16 query rows x full Lk.
// Block = 128 thr (4 waves -> 64 q rows). Grid (Lq/64, H, B).
// All 4 waves share the (b,h) K/V stream: 32x64 K and V tiles are TDM-DMA'd
// into LDS, double-buffered. Q pre-scaled by 1/64; K pre-scaled by dpf.
// ---------------------------------------------------------------------------
__global__ void attn_kernel(const __bf16* __restrict__ Qp,
                            const __bf16* __restrict__ Kb,
                            const __bf16* __restrict__ Vp,
                            const unsigned char* __restrict__ mask,
                            __bf16* __restrict__ Ctx) {
  const int lane = threadIdx.x & 31;
  const int w = threadIdx.x >> 5;
  const int r = lane & 15, hi = lane >> 4;
  const int b = blockIdx.z, h = blockIdx.y;
  const int q0 = blockIdx.x * 64 + w * 16;

  const __bf16* Qh = Qp + (size_t)(b * 16 + h) * 512 * 64;
  const __bf16* Kh = Kb + (size_t)(b * 16 + h) * 1024 * 64;
  const __bf16* Vh = Vp + (size_t)(b * 16 + h) * 1024 * 64;
  const unsigned char* mk = mask + b * 1024;

  __shared__ __bf16 kt[2][32][64];             // [buf][key][dh]
  __shared__ __bf16 vt[2][32][64];             // [buf][key][dh]
  __shared__ __bf16 lds_p[4][16][32];          // per-wave P relayout (C->A)

  if (w == 0) {                                // prologue DMA: kc = 0
    tdm_load_2d((unsigned int)(uintptr_t)&kt[0][0][0], Kh, 64, 64, 32);
    tdm_load_2d((unsigned int)(uintptr_t)&vt[0][0][0], Vh, 64, 64, 32);
  }

  const v16bf a_lo = load_tile(Qh + (size_t)q0 * 64, 64);       // dh 0..31
  const v16bf a_hi = load_tile(Qh + (size_t)q0 * 64 + 32, 64);  // dh 32..63

  v8f acc[4] = {};                             // O tile: 16 x 64 fp32
  float m_run[8], l_run[8];
#pragma unroll
  for (int j = 0; j < 8; ++j) { m_run[j] = -3.0e38f; l_run[j] = 0.f; }

  for (int kc = 0; kc < 1024; kc += 32) {
    const int cur = (kc >> 5) & 1;
    if (w == 0) {
      if (kc + 32 < 1024) {
        tdm_load_2d((unsigned int)(uintptr_t)&kt[cur ^ 1][0][0],
                    Kh + (size_t)(kc + 32) * 64, 64, 64, 32);
        tdm_load_2d((unsigned int)(uintptr_t)&vt[cur ^ 1][0][0],
                    Vh + (size_t)(kc + 32) * 64, 64, 64, 32);
        __builtin_amdgcn_s_wait_tensorcnt(2);  // cur pair landed (in-order)
      } else {
        __builtin_amdgcn_s_wait_tensorcnt(0);
      }
    }
    __syncthreads();                           // publish kt/vt[cur]

    // ---- scores: two 16x16 tiles (keys kc..kc+15, kc+16..kc+31)
    v8f s0 = {}, s1 = {};
    {
      v16bf b0l = load_tile(&kt[cur][0][0], 64);
      v16bf b0h = load_tile(&kt[cur][0][32], 64);
      s0 = wmma_bf16(a_lo, b0l, s0);
      s0 = wmma_bf16(a_hi, b0h, s0);
      v16bf b1l = load_tile(&kt[cur][16][0], 64);
      v16bf b1h = load_tile(&kt[cur][16][32], 64);
      s1 = wmma_bf16(a_lo, b1l, s1);
      s1 = wmma_bf16(a_hi, b1h, s1);
    }
    // ---- mask (lane owns key column kc+r / kc+16+r)
    const bool ok0 = mk[kc + r] != 0;
    const bool ok1 = mk[kc + 16 + r] != 0;
#pragma unroll
    for (int j = 0; j < 8; ++j) {
      if (!ok0) s0[j] = NEG_INF;
      if (!ok1) s1[j] = NEG_INF;
    }
    // ---- online softmax; row (8hi+j) spans the 16 lanes of this half-wave
    float p0[8], p1[8], alpha[8];
#pragma unroll
    for (int j = 0; j < 8; ++j) {
      float vm = fmaxf(s0[j], s1[j]);
      vm = fmaxf(vm, __shfl_xor(vm, 1, 32));
      vm = fmaxf(vm, __shfl_xor(vm, 2, 32));
      vm = fmaxf(vm, __shfl_xor(vm, 4, 32));
      vm = fmaxf(vm, __shfl_xor(vm, 8, 32));
      const float mn = fmaxf(m_run[j], vm);
      alpha[j] = __expf(m_run[j] - mn);
      m_run[j] = mn;
      p0[j] = __expf(s0[j] - mn);
      p1[j] = __expf(s1[j] - mn);
      float rs = p0[j] + p1[j];
      rs += __shfl_xor(rs, 1, 32);
      rs += __shfl_xor(rs, 2, 32);
      rs += __shfl_xor(rs, 4, 32);
      rs += __shfl_xor(rs, 8, 32);
      l_run[j] = l_run[j] * alpha[j] + rs;
    }
#pragma unroll
    for (int t = 0; t < 4; ++t)
#pragma unroll
      for (int j = 0; j < 8; ++j) acc[t][j] *= alpha[j];

    // ---- re-layout P (C layout) -> A layout via LDS
    __syncthreads();
#pragma unroll
    for (int j = 0; j < 8; ++j) {
      lds_p[w][hi * 8 + j][r]      = (__bf16)p0[j];
      lds_p[w][hi * 8 + j][16 + r] = (__bf16)p1[j];
    }
    __syncthreads();
    v16bf pa = load_tile(&lds_p[w][0][0], 32);

    // ---- O += P(16x32) . V(32x64): 4 N-tiles; lane = dh column 16t+r
#pragma unroll
    for (int t = 0; t < 4; ++t) {
      const __bf16* vcol = &vt[cur][0][t * 16 + r];
      v16bf bv;
#pragma unroll
      for (int i = 0; i < 8; ++i) {
        bv[i]     = vcol[(size_t)(hi * 8 + i) * 64];
        bv[8 + i] = vcol[(size_t)(16 + hi * 8 + i) * 64];
      }
      acc[t] = wmma_bf16(pa, bv, acc[t]);
    }
    __syncthreads();                           // done reading kt/vt[cur]
  }

  // ---- normalize + write context in [B, Lq, H, dh] (== [B, Lq, D]) bf16
#pragma unroll
  for (int t = 0; t < 4; ++t)
#pragma unroll
    for (int j = 0; j < 8; ++j) {
      const float inv = 1.0f / l_run[j];
      const int q = q0 + hi * 8 + j;
      const int n = t * 16 + r;
      Ctx[(((size_t)b * 512 + q) * 16 + h) * 64 + n] = (__bf16)(acc[t][j] * inv);
    }
}

// ---------------------------------------------------------------------------
// Host orchestration
// ---------------------------------------------------------------------------
extern "C" void kernel_launch(void* const* d_in, const int* in_sizes, int n_in,
                              void* d_out, int out_size, void* d_ws, size_t ws_size,
                              hipStream_t stream) {
  (void)in_sizes; (void)n_in; (void)out_size; (void)ws_size;
  const float* query = (const float*)d_in[0];   // [4,512,1024]
  const float* key   = (const float*)d_in[1];   // [4,1024,1024]
  const unsigned char* mask = (const unsigned char*)d_in[2]; // [4,1024] bool
  const int* segments = (const int*)d_in[3];    // [4,1024]
  const float* Wq = (const float*)d_in[4];
  const float* bq = (const float*)d_in[5];
  const float* Wk = (const float*)d_in[6];
  const float* bk = (const float*)d_in[7];
  const float* Wv = (const float*)d_in[8];
  const float* bv = (const float*)d_in[9];
  const float* Wo = (const float*)d_in[10];
  const float* bo = (const float*)d_in[11];
  const float* Wb = (const float*)d_in[12];
  const float* bb = (const float*)d_in[13];
  // d_in[14]=n_heads(16), d_in[15]=n_segments(5): baked in as constants.

  const int Mq = 4 * 512;    // 2048
  const int Mk = 4 * 1024;   // 4096
  const int D  = 1024;

  char* ws = (char*)d_ws;
  size_t off = 0;
  auto alloc = [&](size_t bytes) -> void* {
    void* p = ws + off;
    off += bytes;
    off = (off + 255) & ~(size_t)255;
    return p;
  };
  __bf16* qbf  = (__bf16*)alloc((size_t)Mq * D * 2);
  __bf16* kbf  = (__bf16*)alloc((size_t)Mk * D * 2);
  __bf16* wqb  = (__bf16*)alloc((size_t)D * D * 2);
  __bf16* wkb  = (__bf16*)alloc((size_t)D * D * 2);
  __bf16* wvb  = (__bf16*)alloc((size_t)D * D * 2);
  __bf16* wob  = (__bf16*)alloc((size_t)D * D * 2);
  float*  dpf  = (float*)alloc((size_t)Mk * 4);
  __bf16* Qp   = (__bf16*)alloc((size_t)Mq * D * 2);   // [B,H,Lq,dh]
  __bf16* Kbp  = (__bf16*)alloc((size_t)Mk * D * 2);   // [B,H,Lk,dh] (x dpf)
  __bf16* Vpp  = (__bf16*)alloc((size_t)Mk * D * 2);   // [B,H,Lk,dh]
  __bf16* Ctx  = (__bf16*)alloc((size_t)Mq * D * 2);   // [B,Lq,D]

  // 1) fp32 -> bf16 conversions
  cvt_f32_bf16<<<(Mq * D) / 256, 256, 0, stream>>>(query, qbf, Mq * D);
  cvt_f32_bf16<<<(Mk * D) / 256, 256, 0, stream>>>(key, kbf, Mk * D);
  cvt_f32_bf16<<<(D * D) / 256, 256, 0, stream>>>(Wq, wqb, D * D);
  cvt_f32_bf16<<<(D * D) / 256, 256, 0, stream>>>(Wk, wkb, D * D);
  cvt_f32_bf16<<<(D * D) / 256, 256, 0, stream>>>(Wv, wvb, D * D);
  cvt_f32_bf16<<<(D * D) / 256, 256, 0, stream>>>(Wo, wob, D * D);

  // 2) dialogue-position factor (cos term + hardtanh(key . Wb + bb))
  dpf_kernel<<<Mk / 8, 256, 0, stream>>>(key, Wb, bb, segments, dpf);

  // 3) projections (WMMA GEMMs, TDM-staged weights)
  gemm_bf16<<<dim3(Mq / 128, D / 64), 256, 0, stream>>>(
      qbf, wqb, bq, nullptr, 1.0f / 64.0f, Qp, nullptr, Mq, D, D, 1, 512);
  gemm_bf16<<<dim3(Mk / 128, D / 64), 256, 0, stream>>>(
      kbf, wkb, bk, dpf, 0.0f, Kbp, nullptr, Mk, D, D, 1, 1024);
  gemm_bf16<<<dim3(Mk / 128, D / 64), 256, 0, stream>>>(
      kbf, wvb, bv, nullptr, 1.0f, Vpp, nullptr, Mk, D, D, 1, 1024);

  // 4) flash attention (TDM-staged K/V tiles)
  attn_kernel<<<dim3(512 / 64, 16, 4), 128, 0, stream>>>(Qp, Kbp, Vpp, mask, Ctx);

  // 5) output projection -> fp32 d_out
  gemm_bf16<<<dim3(Mq / 128, D / 64), 256, 0, stream>>>(
      Ctx, wob, bo, nullptr, 1.0f, nullptr, (float*)d_out, Mq, D, D, 0, 1);
}
